// ResNet_79078937854186
// MI455X (gfx1250) — matrix operand order repaired and checked by reading, not compiled, not run on hardware
//
#include <hip/hip_runtime.h>
#include <hip/hip_bf16.h>
#include <stdint.h>

// ---------------------------------------------------------------------------
// SplineConv ResNet forward for MI455X (gfx1250).
// Dense GEMMs -> v_wmma_f32_16x16x32_bf16, LDS staged via async global->LDS
// DMA (ASYNCcnt) when the toolchain exposes it, else b128 vector copies.
// ---------------------------------------------------------------------------

#define NN        2048      // nodes
#define EE        32768     // edges
#define NB        8         // graphs in batch
#define POOLSZ    4
#define CMAX      512
#define NCOLS_MAX 5120      // (9+1)*512
#define EPSBN     1e-5f

typedef __attribute__((ext_vector_type(16))) __bf16 v16bf;
typedef __attribute__((ext_vector_type(8)))  __bf16 v8bf;
typedef __attribute__((ext_vector_type(8)))  float  v8f;
typedef int v4i_ __attribute__((vector_size(16)));   // matches builtin param

#if defined(__has_builtin)
#if __has_builtin(__builtin_amdgcn_global_load_async_to_lds_b128) && \
    __has_builtin(__builtin_amdgcn_s_wait_asynccnt)
#define HAVE_ASYNC_LDS 1
#endif
#endif

#define ASG __attribute__((address_space(1)))
#define ASL __attribute__((address_space(3)))

__device__ __forceinline__ __bf16 f2bf(float f) {
    unsigned u = __float_as_uint(f);
    unsigned r = u + 0x7FFFu + ((u >> 16) & 1u);      // round-to-nearest-even
    unsigned short hs = (unsigned short)(r >> 16);
    __bf16 b;
    __builtin_memcpy(&b, &hs, 2);
    return b;
}

// ------------------------------- packing -----------------------------------

// K-major (transposed) packed weights: WbT[c, i] with c = k*cout+o (k<K2) or
// c = K2*cout+o (root).  Row length cinP (padded), so each output column's
// K-run is contiguous -> B fragments become two ds_load_b128 per lane.
__global__ void pack_w_kernel(const float* __restrict__ W,
                              const float* __restrict__ root,
                              __bf16* __restrict__ WbT,
                              int cin, int cout, int K2, int cinP, int ncols) {
    int idx = blockIdx.x * blockDim.x + threadIdx.x;
    int total = ncols * cinP;
    if (idx >= total) return;
    int c = idx / cinP, i = idx % cinP;
    float v = 0.0f;
    if (i < cin) {
        int k = c / cout, o = c % cout;
        v = (k < K2) ? W[((size_t)k * cin + i) * cout + o]
                     : root[(size_t)i * cout + o];
    }
    WbT[idx] = f2bf(v);
}

__global__ void pack_a_kernel(const float* __restrict__ x,
                              __bf16* __restrict__ A,
                              int cin, int cinP) {
    int idx = blockIdx.x * blockDim.x + threadIdx.x;
    int total = NN * cinP;
    if (idx >= total) return;
    int n = idx / cinP, c = idx % cinP;
    A[idx] = f2bf(c < cin ? x[(size_t)n * cin + c] : 0.0f);
}

// ------------------------------- WMMA GEMM ---------------------------------
// Y[NN, ncols] = A[NN, Kdim](bf16) * BwT[ncols, Kdim](bf16)^T, f32 accumulate.
// Block tile 64(M) x 128(N); 8 waves = 4(M) x 2(N-groups); each wave owns
// four 16x16 WMMA accumulators (one shared A fragment per K-slab).

#define TM 64
#define TN 128
#define TK 32
#define AS_STRIDE 5   // uint4 per LDS row (80B) -> 16B aligned, conflict-free
#define BS_STRIDE 5

__global__ void __launch_bounds__(256)
gemm_wmma_kernel(const __bf16* __restrict__ A, const __bf16* __restrict__ BwT,
                 float* __restrict__ Y, int Kdim, int ncols) {
    __shared__ uint4 As4[TM * AS_STRIDE];   // 64 rows x 32 K  (bf16)
    __shared__ uint4 Bs4[TN * BS_STRIDE];   // 128 cols x 32 K (bf16, K-major)

    const int m0 = blockIdx.x * TM;
    const int n0 = blockIdx.y * TN;
    const int tid = threadIdx.x;
    const int w = tid >> 5, lane = tid & 31;
    const int mi = w & 3;                   // M sub-tile (16 rows)
    const int ni = w >> 2;                  // N group (64 cols)
    const int h = lane >> 4, lr = lane & 15;

    const uint4* Ag = (const uint4*)A;
    const uint4* Bg = (const uint4*)BwT;
    const int Ku4 = Kdim >> 3;              // global row stride in uint4

    const int ar = tid >> 2, aq = tid & 3;          // A stage coords
    const int bc0 = tid >> 2, bq0 = tid & 3;        // B stage coords (first)
    const int bc1 = (tid + 256) >> 2, bq1 = tid & 3;

    uint4* adst = &As4[ar * AS_STRIDE + aq];
    uint4* bdst0 = &Bs4[bc0 * BS_STRIDE + bq0];
    uint4* bdst1 = &Bs4[bc1 * BS_STRIDE + bq1];

    v8f acc[4] = {};

    for (int k0 = 0; k0 < Kdim; k0 += TK) {
        const int kq = k0 >> 3;
        const uint4* asrc = Ag + (size_t)(m0 + ar) * Ku4 + kq + aq;
        const uint4* bsrc0 = Bg + (size_t)(n0 + bc0) * Ku4 + kq + bq0;
        const uint4* bsrc1 = Bg + (size_t)(n0 + bc1) * Ku4 + kq + bq1;

#if HAVE_ASYNC_LDS
        // gfx1250 async DMA: memory -> LDS, no VGPR roundtrip (ASYNCcnt).
        __builtin_amdgcn_global_load_async_to_lds_b128(
            (ASG v4i_*)(uintptr_t)asrc,
            (ASL v4i_*)(uint32_t)(uintptr_t)adst, 0, 0);
        __builtin_amdgcn_global_load_async_to_lds_b128(
            (ASG v4i_*)(uintptr_t)bsrc0,
            (ASL v4i_*)(uint32_t)(uintptr_t)bdst0, 0, 0);
        __builtin_amdgcn_global_load_async_to_lds_b128(
            (ASG v4i_*)(uintptr_t)bsrc1,
            (ASL v4i_*)(uint32_t)(uintptr_t)bdst1, 0, 0);
#else
        *adst  = *asrc;
        *bdst0 = *bsrc0;
        *bdst1 = *bsrc1;
#endif
        if (k0 + TK < Kdim) {               // speculative prefetch of next slab
            __builtin_prefetch(asrc + 4, 0, 1);
            __builtin_prefetch(bsrc0 + 4, 0, 1);
        }
#if HAVE_ASYNC_LDS
        __builtin_amdgcn_s_wait_asynccnt(0);
#endif
        __syncthreads();

        // ---- A fragment: lane = row (mi*16+lr); two contiguous 8-elem K runs
        //      K = e + 8h (e<8)  and  K = e + 8 + 8h (e>=8)
        const v8bf* Asb = (const v8bf*)As4;
        v8bf alo = Asb[(mi * 16 + lr) * AS_STRIDE + h];
        v8bf ahi = Asb[(mi * 16 + lr) * AS_STRIDE + 2 + h];
        v16bf afrag = __builtin_shufflevector(alo, ahi,
            0, 1, 2, 3, 4, 5, 6, 7, 8, 9, 10, 11, 12, 13, 14, 15);

        // ---- 4 WMMAs: B fragment = two 8-elem K runs of one column (K-major), same K order as A
        const v8bf* Bsb = (const v8bf*)Bs4;
#pragma unroll
        for (int t = 0; t < 4; ++t) {
            int col = ni * 64 + t * 16 + lr;
            v8bf blo = Bsb[col * BS_STRIDE + h];
            v8bf bhi = Bsb[col * BS_STRIDE + 2 + h];
            v16bf bfrag = __builtin_shufflevector(blo, bhi,
                0, 1, 2, 3, 4, 5, 6, 7, 8, 9, 10, 11, 12, 13, 14, 15);
            acc[t] = __builtin_amdgcn_wmma_f32_16x16x32_bf16(
                /*neg_a=*/false, afrag, /*neg_b=*/false, bfrag,
                /*c_mod=*/(short)0, acc[t], /*reuse_a=*/false, /*reuse_b=*/false);
        }
        __syncthreads();
    }

    // C/D layout: VGPR r -> M = r (lanes 0-15) / r+8 (lanes 16-31), N = lane%16
#pragma unroll
    for (int t = 0; t < 4; ++t) {
        int col = n0 + ni * 64 + t * 16 + lr;
#pragma unroll
        for (int r = 0; r < 8; ++r) {
            int row = m0 + mi * 16 + r + h * 8;
            Y[(size_t)row * ncols + col] = acc[t][r];
        }
    }
}

// ------------------------------ graph ops ----------------------------------

__global__ void deg_kernel(const int* __restrict__ ei, float* __restrict__ deg) {
    int e = blockIdx.x * blockDim.x + threadIdx.x;
    if (e < EE) atomicAdd(&deg[ei[EE + e]], 1.0f);
}

// one wave per edge: fold 4 bilinear B-spline basis terms into a register sum,
// then a single global_atomic_add_f32 per channel.
__global__ void __launch_bounds__(256)
edge_kernel(const int* __restrict__ ei, const float* __restrict__ ea,
            const float* __restrict__ Y, float* __restrict__ agg,
            int cout, int ncols, int K) {
    int gw = (blockIdx.x * blockDim.x + threadIdx.x) >> 5;  // edge id
    int lane = threadIdx.x & 31;
    if (gw >= EE) return;
    int src = ei[gw];
    int dst = ei[EE + gw];
    float u = ea[2 * gw + 0], v = ea[2 * gw + 1];
    float vx = u * (float)(K - 1), vy = v * (float)(K - 1);
    int hiK = (K - 2 > 0) ? (K - 2) : 0;
    int lox = min(max((int)floorf(vx), 0), hiK);
    int loy = min(max((int)floorf(vy), 0), hiK);
    float fx = vx - (float)lox, fy = vy - (float)loy;

    float wgt[4];
    int   koff[4];
#pragma unroll
    for (int bb = 0; bb < 4; ++bb) {
        int b0 = bb >> 1, b1 = bb & 1;
        int i0 = min(lox + b0, K - 1);
        int i1 = min(loy + b1, K - 1);
        wgt[bb]  = (b0 ? fx : 1.0f - fx) * (b1 ? fy : 1.0f - fy);
        koff[bb] = (i0 * K + i1) * cout;
    }

    const float* ybase = Y + (size_t)src * ncols;
    float* arow = agg + (size_t)dst * cout;
    for (int o = lane; o < cout; o += 32) {
        float s = wgt[0] * ybase[koff[0] + o] + wgt[1] * ybase[koff[1] + o]
                + wgt[2] * ybase[koff[2] + o] + wgt[3] * ybase[koff[3] + o];
        atomicAdd(&arow[o], s);
    }
}

// out = agg/max(deg,1) + root-product (last cout cols of Y) + bias
__global__ void combine_kernel(const float* __restrict__ Y,
                               const float* __restrict__ agg,
                               const float* __restrict__ deg,
                               const float* __restrict__ bias,
                               float* __restrict__ out, int cout, int ncols) {
    int idx = blockIdx.x * blockDim.x + threadIdx.x;
    if (idx >= NN * cout) return;
    int n = idx / cout, o = idx % cout;
    float d = fmaxf(deg[n], 1.0f);
    out[idx] = agg[idx] / d + Y[(size_t)n * ncols + (ncols - cout) + o] + bias[o];
}

// ------------------------------ batchnorm ----------------------------------

__global__ void __launch_bounds__(256)
bn_stats_kernel(const float* __restrict__ h, int C, float* __restrict__ stats) {
    int c = blockIdx.x;
    float s = 0.0f, s2 = 0.0f;
    for (int n = threadIdx.x; n < NN; n += 256) {
        float v = h[(size_t)n * C + c];
        s += v; s2 += v * v;
    }
    __shared__ float sh[256], sh2[256];
    sh[threadIdx.x] = s; sh2[threadIdx.x] = s2;
    __syncthreads();
    for (int st = 128; st > 0; st >>= 1) {
        if (threadIdx.x < st) {
            sh[threadIdx.x]  += sh[threadIdx.x + st];
            sh2[threadIdx.x] += sh2[threadIdx.x + st];
        }
        __syncthreads();
    }
    if (threadIdx.x == 0) {
        float m = sh[0] / (float)NN;
        float var = sh2[0] / (float)NN - m * m;
        stats[2 * c]     = m;
        stats[2 * c + 1] = var;
    }
}

__global__ void bn_apply_kernel(const float* __restrict__ h,
                                const float* __restrict__ stats,
                                const float* __restrict__ gamma,
                                const float* __restrict__ beta,
                                const float* __restrict__ resid,  // may be null
                                float* __restrict__ out, int C, int do_relu) {
    int idx = blockIdx.x * blockDim.x + threadIdx.x;
    if (idx >= NN * C) return;
    int c = idx % C;
    float m = stats[2 * c], v = stats[2 * c + 1];
    float y = gamma[c] * (h[idx] - m) * rsqrtf(v + EPSBN) + beta[c];
    if (resid) y += resid[idx];
    if (do_relu) y = fmaxf(y, 0.0f);
    out[idx] = y;
}

// ------------------------------ pool + fc ----------------------------------

__global__ void pool_kernel(const float* __restrict__ h,
                            const float* __restrict__ pos,
                            const int* __restrict__ batch,
                            float* __restrict__ pooled, int C) {
    int idx = blockIdx.x * blockDim.x + threadIdx.x;
    if (idx >= NN * C) return;
    int n = idx / C, c = idx % C;
    int c0 = (int)floorf(pos[2 * n + 0] * 0.25f);
    int c1 = (int)floorf(pos[2 * n + 1] * 0.25f);
    int lin = c0 * 4096 + c1;
    int slot = batch[n] * POOLSZ + (lin & (POOLSZ - 1));
    float v = fmaxf(h[idx], 0.0f);  // h already ReLU'd; int-ordering valid
    atomicMax((int*)&pooled[(size_t)slot * C + c], __float_as_int(v));
}

__global__ void fc_kernel(const float* __restrict__ pooled,
                          const float* __restrict__ w,
                          const float* __restrict__ b,
                          float* __restrict__ out) {
    int idx = blockIdx.x * blockDim.x + threadIdx.x;
    if (idx >= NB * POOLSZ * 10) return;
    int i = idx / 10, j = idx % 10;
    float s = b[j];
    for (int k = 0; k < 512; ++k)
        s += pooled[(size_t)i * 512 + k] * w[(size_t)k * 10 + j];
    out[idx] = s;
}

// ------------------------------ host side ----------------------------------

struct ConvP { const float *W, *root, *bias; };
struct BNP   { const float *gamma, *beta; };

extern "C" void kernel_launch(void* const* d_in, const int* in_sizes, int n_in,
                              void* d_out, int out_size, void* d_ws, size_t ws_size,
                              hipStream_t stream) {
    (void)in_sizes; (void)n_in; (void)out_size; (void)ws_size;

    const float* x    = (const float*)d_in[0];
    const int*   ei   = (const int*)  d_in[1];
    const float* ea   = (const float*)d_in[2];
    const float* pos  = (const float*)d_in[3];
    const int*   batch= (const int*)  d_in[4];

    int pi = 5;
    auto nextConv = [&](ConvP& p) {
        p.W    = (const float*)d_in[pi++];
        p.root = (const float*)d_in[pi++];
        p.bias = (const float*)d_in[pi++];
    };
    auto nextBN = [&](BNP& p) {
        p.gamma = (const float*)d_in[pi++];
        p.beta  = (const float*)d_in[pi++];
    };

    ConvP conv1;            BNP bn1;
    nextConv(conv1);        nextBN(bn1);
    ConvP l1c1, l1c2;       BNP l1b1, l1b2;
    nextConv(l1c1); nextBN(l1b1); nextConv(l1c2); nextBN(l1b2);
    ConvP l2c1, l2c2, l2dc; BNP l2b1, l2b2, l2dbn;
    nextConv(l2c1); nextBN(l2b1); nextConv(l2c2); nextBN(l2b2);
    nextConv(l2dc); nextBN(l2dbn);
    ConvP l3c1, l3c2, l3dc; BNP l3b1, l3b2, l3dbn;
    nextConv(l3c1); nextBN(l3b1); nextConv(l3c2); nextBN(l3b2);
    nextConv(l3dc); nextBN(l3dbn);
    ConvP l4c1, l4c2, l4dc; BNP l4b1, l4b2, l4dbn;
    nextConv(l4c1); nextBN(l4b1); nextConv(l4c2); nextBN(l4b2);
    nextConv(l4dc); nextBN(l4dbn);
    const float* fc_w = (const float*)d_in[pi++];
    const float* fc_b = (const float*)d_in[pi++];

    // ---- workspace carve-up (256B aligned) ----
    char*  base = (char*)d_ws;
    size_t off  = 0;
    auto carve = [&](size_t bytes) -> char* {
        char* p = base + off;
        off += (bytes + 255) & ~(size_t)255;
        return p;
    };
    float*  deg    = (float*) carve(NN * sizeof(float));
    float*  buf0   = (float*) carve((size_t)NN * CMAX * sizeof(float));
    float*  buf1   = (float*) carve((size_t)NN * CMAX * sizeof(float));
    float*  buf2   = (float*) carve((size_t)NN * CMAX * sizeof(float));
    __bf16* Abf    = (__bf16*)carve((size_t)NN * CMAX * sizeof(__bf16));
    __bf16* WbT    = (__bf16*)carve((size_t)CMAX * NCOLS_MAX * sizeof(__bf16));
    float*  Y      = (float*) carve((size_t)NN * NCOLS_MAX * sizeof(float));
    float*  agg    = (float*) carve((size_t)NN * CMAX * sizeof(float));
    float*  stats  = (float*) carve(2 * CMAX * sizeof(float));
    float*  pooled = (float*) carve((size_t)NB * POOLSZ * CMAX * sizeof(float));

    // degree histogram (edge-structure constant across layers)
    (void)hipMemsetAsync(deg, 0, NN * sizeof(float), stream);
    deg_kernel<<<(EE + 255) / 256, 256, 0, stream>>>(ei, deg);

    auto spline = [&](const float* xin, const ConvP& cp, int cin, int cout,
                      int K, float* out) {
        int K2    = K * K;
        int cinP  = (cin + 31) & ~31;
        int ncols = (K2 + 1) * cout;       // always a multiple of 128 here

        int wtot = ncols * cinP;
        pack_w_kernel<<<(wtot + 255) / 256, 256, 0, stream>>>(
            cp.W, cp.root, WbT, cin, cout, K2, cinP, ncols);
        int atot = NN * cinP;
        pack_a_kernel<<<(atot + 255) / 256, 256, 0, stream>>>(xin, Abf, cin, cinP);

        dim3 g(NN / TM, ncols / TN);
        gemm_wmma_kernel<<<g, 256, 0, stream>>>(Abf, WbT, Y, cinP, ncols);

        (void)hipMemsetAsync(agg, 0, (size_t)NN * cout * sizeof(float), stream);
        edge_kernel<<<EE / 8, 256, 0, stream>>>(ei, ea, Y, agg, cout, ncols, K);

        combine_kernel<<<(NN * cout + 255) / 256, 256, 0, stream>>>(
            Y, agg, deg, cp.bias, out, cout, ncols);
    };

    auto bn = [&](const float* h, const BNP& p, const float* resid,
                  int relu, float* out, int C) {
        bn_stats_kernel<<<C, 256, 0, stream>>>(h, C, stats);
        bn_apply_kernel<<<(NN * C + 255) / 256, 256, 0, stream>>>(
            h, stats, p.gamma, p.beta, resid, out, C, relu);
    };

    // ---- forward pass ----
    // stem: h = relu(bn(conv1(x)))
    spline(x, conv1, 3, 64, 3, buf0);
    bn(buf0, bn1, nullptr, 1, buf1, 64);

    // layer1 (64 -> 64, identity residual)
    spline(buf1, l1c1, 64, 64, 3, buf0);
    bn(buf0, l1b1, nullptr, 1, buf2, 64);
    spline(buf2, l1c2, 64, 64, 3, buf0);
    bn(buf0, l1b2, buf1, 1, buf1, 64);

    // layer2 (64 -> 128, dim-change residual)
    spline(buf1, l2c1, 64, 128, 3, buf0);
    bn(buf0, l2b1, nullptr, 1, buf2, 128);
    spline(buf2, l2c2, 128, 128, 3, buf0);
    spline(buf1, l2dc, 64, 128, 1, buf2);
    bn(buf2, l2dbn, nullptr, 0, buf2, 128);
    bn(buf0, l2b2, buf2, 1, buf1, 128);

    // layer3 (128 -> 256)
    spline(buf1, l3c1, 128, 256, 3, buf0);
    bn(buf0, l3b1, nullptr, 1, buf2, 256);
    spline(buf2, l3c2, 256, 256, 3, buf0);
    spline(buf1, l3dc, 128, 256, 1, buf2);
    bn(buf2, l3dbn, nullptr, 0, buf2, 256);
    bn(buf0, l3b2, buf2, 1, buf1, 256);

    // layer4 (256 -> 512)
    spline(buf1, l4c1, 256, 512, 3, buf0);
    bn(buf0, l4b1, nullptr, 1, buf2, 512);
    spline(buf2, l4c2, 512, 512, 3, buf0);
    spline(buf1, l4dc, 256, 512, 1, buf2);
    bn(buf2, l4dbn, nullptr, 0, buf2, 512);
    bn(buf0, l4b2, buf2, 1, buf1, 512);

    // voxel max-pool + fc
    (void)hipMemsetAsync(pooled, 0, (size_t)NB * POOLSZ * 512 * sizeof(float), stream);
    pool_kernel<<<(NN * 512 + 255) / 256, 256, 0, stream>>>(buf1, pos, batch,
                                                            pooled, 512);
    fc_kernel<<<2, 256, 0, stream>>>(pooled, fc_w, fc_b, (float*)d_out);
}
